// BoundaryAttention_54228257079960
// MI455X (gfx1250) — matrix operand order repaired and checked by reading, not compiled, not run on hardware
//
#include <hip/hip_runtime.h>
#include <hip/hip_bf16.h>
#include <math.h>

// ---------------------------------------------------------------------------
// MI455X (gfx1250) implementation.
//   * wave32, WMMA 16x16x32 f16 (f32 accum) for all GEMMs and attention.
//   * Whole problem (~16 MB activations) is L2-resident (192 MB), so global
//     loads per-wave are fine; LDS is used only where a transposed operand is
//     required (V for the PV matmul, P for the A-operand relayout).
// ---------------------------------------------------------------------------

typedef _Float16 h8   __attribute__((ext_vector_type(8)));
typedef _Float16 v16h __attribute__((ext_vector_type(16)));
typedef float    v8f  __attribute__((ext_vector_type(8)));

#define NPIX  16384      // 128*128
#define DPAD  96         // padded feature stride (f16 elements)
#define IH    128
#define IW    128

static __device__ __forceinline__ v16h pack16(h8 lo, h8 hi) {
  v16h r;
#pragma unroll
  for (int i = 0; i < 8; ++i) { r[i] = lo[i]; r[i + 8] = hi[i]; }
  return r;
}

static __device__ __forceinline__ int iclamp(int v, int lo, int hi) {
  return v < lo ? lo : (v > hi ? hi : v);
}

// ---------------------------------------------------------------------------
// Weight prep: transpose+pad 8 weight matrices to WT[96][96] f16 (WT[n][k]),
// pad biases to 96 f32.  mat order: 0=mlp1 1=mlp2 2=wq1 3=wk1 4=wv1 5=wq2
// 6=wk2 7=wv2.  K_real = 67 for mlp2, else 72.  N_real = 72 for all.
// ---------------------------------------------------------------------------
__global__ void prep_weights_kernel(
    const float* w0, const float* w1, const float* w2, const float* w3,
    const float* w4, const float* w5, const float* w6, const float* w7,
    const float* b0, const float* b1, const float* b2, const float* b3,
    const float* b4, const float* b5, const float* b6, const float* b7,
    _Float16* __restrict__ wt, float* __restrict__ bias)
{
  const int idx = blockIdx.x * blockDim.x + threadIdx.x;
  const int WT_TOT = 8 * 96 * 96;
  if (idx < WT_TOT) {
    int mat = idx / 9216;
    int rem = idx - mat * 9216;
    int n = rem / 96, k = rem - n * 96;
    int kr = (mat == 1) ? 67 : 72;
    const float* w;
    switch (mat) {
      case 0: w = w0; break; case 1: w = w1; break;
      case 2: w = w2; break; case 3: w = w3; break;
      case 4: w = w4; break; case 5: w = w5; break;
      case 6: w = w6; break; default: w = w7; break;
    }
    float v = (k < kr && n < 72) ? w[k * 72 + n] : 0.0f;
    wt[idx] = (_Float16)v;
  } else if (idx < WT_TOT + 8 * 96) {
    int j = idx - WT_TOT;
    int mat = j / 96, n = j - mat * 96;
    const float* b;
    switch (mat) {
      case 0: b = b0; break; case 1: b = b1; break;
      case 2: b = b2; break; case 3: b = b3; break;
      case 4: b = b4; break; case 5: b = b5; break;
      case 6: b = b6; break; default: b = b7; break;
    }
    bias[j] = (n < 72) ? b[n] : 0.0f;
  }
}

// ---------------------------------------------------------------------------
// Input prep: X1 = [gamma0+gammat | pi | 0] (96), X2 = [gamma | f | 0] (96),
// both f16 [NPIX][96].
// ---------------------------------------------------------------------------
__global__ void prep_inputs_kernel(
    const float* __restrict__ f_t, const float* __restrict__ g0,
    const float* __restrict__ gt, const float* __restrict__ pi_t,
    _Float16* __restrict__ X1, _Float16* __restrict__ X2)
{
  const int idx = blockIdx.x * blockDim.x + threadIdx.x;
  if (idx >= 2 * NPIX * DPAD) return;
  int which = idx / (NPIX * DPAD);
  int rem = idx - which * NPIX * DPAD;
  int p = rem / DPAD, c = rem - p * DPAD;
  float v = 0.0f;
  if (c < 64) {
    v = g0[p * 64 + c] + gt[p * 64 + c];
  } else if (which == 0) {
    if (c < 72) v = pi_t[p * 8 + (c - 64)];
  } else {
    if (c < 67) v = f_t[p * 3 + (c - 64)];
  }
  (which ? X2 : X1)[rem] = (_Float16)v;
}

// ---------------------------------------------------------------------------
// GEMM: Y[P][96] = act(X[P][96] @ W[96][96(pad)] + bias), f16 in/out, f32 acc.
// One 16-row M-tile per wave; 6 N-tiles x 3 K-chunks = 18 WMMAs per wave.
// WT is pre-transposed so B-operand lanes load two contiguous 16-byte runs.
// ---------------------------------------------------------------------------
__global__ __launch_bounds__(128) void gemm_kernel(
    const _Float16* __restrict__ X, const _Float16* __restrict__ WT,
    const float* __restrict__ bias, _Float16* __restrict__ Y, int act)
{
  const int wave = threadIdx.x >> 5;
  const int lane = threadIdx.x & 31;
  const int tile = blockIdx.x * 4 + wave;
  const int mcol = lane & 15;
  const int hi   = (lane >> 4) & 1;

  // A fragments: lane<16 -> row=lane, K = kb+{0..7,16..23}; lane>=16 -> +8.
  v16h a[3];
#pragma unroll
  for (int c = 0; c < 3; ++c) {
    const _Float16* p = X + (size_t)(tile * 16 + mcol) * DPAD + c * 32 + hi * 8;
    a[c] = pack16(*(const h8*)p, *(const h8*)(p + 16));
  }

  v8f acc[6];
#pragma unroll
  for (int nt = 0; nt < 6; ++nt) {
    v8f c8 = {};
#pragma unroll
    for (int c = 0; c < 3; ++c) {
      // B: column n = mcol, K rows kb + hi*8 + {0..7,16..23} (matches A).
      const _Float16* p = WT + (size_t)(nt * 16 + mcol) * DPAD + c * 32 + hi * 8;
      v16h b = pack16(*(const h8*)p, *(const h8*)(p + 16));
      c8 = __builtin_amdgcn_wmma_f32_16x16x32_f16(false, a[c], false, b,
                                                  (short)0, c8, false, false);
    }
    acc[nt] = c8;
  }

#pragma unroll
  for (int nt = 0; nt < 6; ++nt) {
    int col = nt * 16 + mcol;
    float bv = bias[col];
#pragma unroll
    for (int r = 0; r < 8; ++r) {
      int m = r + hi * 8;                 // C layout: row = r + 8*(lane>=16)
      float v = acc[nt][r] + bv;
      if (act) v = 0.5f * v * (1.0f + erff(v * 0.70710678118654752f)); // exact gelu
      Y[(size_t)(tile * 16 + m) * DPAD + col] = (_Float16)v;
    }
  }
}

// ---------------------------------------------------------------------------
// Neighborhood attention (K=11). One wave handles a 2x8 query tile.
// Union key window: 12x18 = 216 keys -> 14 score tiles (224 padded),
// 7 K-chunks for PV.  Scores: 42 WMMAs (K rows read from global/L2).
// Softmax in registers via width-16 shuffles.  P -> LDS (A layout reads),
// V transposed into LDS -> PV: 35 WMMAs.
// mode 0: write f16 [P][96] (cols 80..95 zeroed).  mode 1: write f32
// gamma_next [P][64] and pi_next [P][8].
// ---------------------------------------------------------------------------
__global__ __launch_bounds__(32) void natt_kernel(
    const _Float16* __restrict__ QH, const _Float16* __restrict__ Km,
    const _Float16* __restrict__ Vm, _Float16* __restrict__ O16,
    float* __restrict__ Og, float* __restrict__ Op, int mode)
{
  __shared__ h8 Vt8[80][28];   // V transposed: [vdim][neighbor/8], 224 cols
  __shared__ h8 P8[16][28];    // probabilities: [query][neighbor/8]
  _Float16* VtH = (_Float16*)Vt8;
  _Float16* PH  = (_Float16*)P8;

  const int lane = threadIdx.x;
  const int mcol = lane & 15;
  const int hi   = lane >> 4;
  const int T  = blockIdx.x;          // 1024 tiles
  const int ty = T >> 4, tx = T & 15;
  const int qy0 = ty * 2, qx0 = tx * 8;
  const int y0 = qy0 - 5, x0 = qx0 - 5;

  __builtin_prefetch(QH + (size_t)(qy0 * IW + qx0) * DPAD, 0, 3);

  // ---- stage V window transposed into LDS -----------------------------
  for (int i = lane; i < 216 * 5; i += 32) {
    int n = i / 5, g = i - n * 5;               // neighbor, 16-dim group
    int wy = n / 18, wx = n - wy * 18;
    int ky = iclamp(y0 + wy, 0, IH - 1);
    int kx = iclamp(x0 + wx, 0, IW - 1);
    const _Float16* vp = Vm + (size_t)(ky * IW + kx) * DPAD + g * 16;
    h8 lo = *(const h8*)vp;
    h8 hh = *(const h8*)(vp + 8);
#pragma unroll
    for (int j = 0; j < 8; ++j) {
      VtH[(g * 16 + j) * 224 + n]     = lo[j];
      VtH[(g * 16 + 8 + j) * 224 + n] = hh[j];
    }
  }

  // ---- Q fragments (A operand), 3 K-chunks ----------------------------
  const int qym = qy0 + (mcol >> 3), qxm = qx0 + (mcol & 7);
  const size_t qrow = (size_t)(qym * IW + qxm) * DPAD;
  v16h aq[3];
#pragma unroll
  for (int c = 0; c < 3; ++c) {
    const _Float16* p = QH + qrow + c * 32 + hi * 8;
    aq[c] = pack16(*(const h8*)p, *(const h8*)(p + 16));
  }

  // ---- scores S = Q K^T : 14 tiles x 3 chunks -------------------------
  v8f acc[14];
#pragma unroll
  for (int t = 0; t < 14; ++t) {
    int nb = t * 16 + mcol;
    int nbc = nb < 216 ? nb : 215;              // clamp pad columns (masked)
    int wy = nbc / 18, wx = nbc - wy * 18;
    int ky = iclamp(y0 + wy, 0, IH - 1);
    int kx = iclamp(x0 + wx, 0, IW - 1);
    const _Float16* kbase = Km + (size_t)(ky * IW + kx) * DPAD + hi * 8;
    v8f c8 = {};
#pragma unroll
    for (int c = 0; c < 3; ++c) {
      v16h b = pack16(*(const h8*)(kbase + c * 32),
                      *(const h8*)(kbase + c * 32 + 16));
      c8 = __builtin_amdgcn_wmma_f32_16x16x32_f16(false, aq[c], false, b,
                                                  (short)0, c8, false, false);
    }
    acc[t] = c8;
  }

  // ---- mask + scale, row max -----------------------------------------
  const float scale = 0.11785113019775793f;     // 1/sqrt(72)
  float rmax[8];
#pragma unroll
  for (int r = 0; r < 8; ++r) rmax[r] = -3.0e38f;

#pragma unroll
  for (int t = 0; t < 14; ++t) {
    int nb = t * 16 + mcol;
    int wy = nb / 18, wx = nb - wy * 18;
    bool inb = (nb < 216) && (y0 + wy >= 0) && (y0 + wy < IH) &&
               (x0 + wx >= 0) && (x0 + wx < IW);
#pragma unroll
    for (int r = 0; r < 8; ++r) {
      int m = r + hi * 8;
      int my = m >> 3, mx = m & 7;
      bool ok = inb && (wy >= my) && (wy <= my + 10) &&
                       (wx >= mx) && (wx <= mx + 10);
      float s = ok ? acc[t][r] * scale : -3.0e38f;
      acc[t][r] = s;
      rmax[r] = fmaxf(rmax[r], s);
    }
  }
#pragma unroll
  for (int r = 0; r < 8; ++r) {
#pragma unroll
    for (int w = 1; w < 16; w <<= 1)
      rmax[r] = fmaxf(rmax[r], __shfl_xor(rmax[r], w, 16));
  }

  // ---- exp, row sum, store P to LDS ----------------------------------
  float rsum[8] = {0.f, 0.f, 0.f, 0.f, 0.f, 0.f, 0.f, 0.f};
#pragma unroll
  for (int t = 0; t < 14; ++t) {
#pragma unroll
    for (int r = 0; r < 8; ++r) {
      float p = expf(acc[t][r] - rmax[r]);      // masked -> exp(-huge) = 0
      rsum[r] += p;
      PH[(r + hi * 8) * 224 + t * 16 + mcol] = (_Float16)p;
    }
  }
#pragma unroll
  for (int r = 0; r < 8; ++r) {
#pragma unroll
    for (int w = 1; w < 16; w <<= 1)
      rsum[r] += __shfl_xor(rsum[r], w, 16);
  }

  __syncthreads();   // Vt + P visible (single wave; ordering fence)

  // ---- O = P V : 7 chunks x 5 N-tiles --------------------------------
  v8f out[5];
#pragma unroll
  for (int nt = 0; nt < 5; ++nt) out[nt] = (v8f){};
#pragma unroll
  for (int c = 0; c < 7; ++c) {
    const h8* prow = &P8[mcol][c * 4 + hi];     // A: K = c*32 + hi*8 {+16}
    v16h pa = pack16(prow[0], prow[2]);
#pragma unroll
    for (int nt = 0; nt < 5; ++nt) {
      const h8* vrow = &Vt8[nt * 16 + mcol][c * 4 + hi];  // B: K = c*32 + hi*8 {+16}
      v16h vb = pack16(vrow[0], vrow[2]);
      out[nt] = __builtin_amdgcn_wmma_f32_16x16x32_f16(false, pa, false, vb,
                                                       (short)0, out[nt],
                                                       false, false);
    }
  }

  // ---- normalize + store ---------------------------------------------
#pragma unroll
  for (int nt = 0; nt < 5; ++nt) {
    int col = nt * 16 + mcol;
#pragma unroll
    for (int r = 0; r < 8; ++r) {
      int m = r + hi * 8;
      int oy = qy0 + (m >> 3), ox = qx0 + (m & 7);
      size_t opix = (size_t)(oy * IW + ox);
      float v = out[nt][r] * (1.0f / rsum[r]);
      if (mode == 0) {
        O16[opix * DPAD + col] = (_Float16)v;
      } else {
        if (col < 64)      Og[opix * 64 + col] = v;
        else if (col < 72) Op[opix * 8 + (col - 64)] = v;
      }
    }
  }
  if (mode == 0) {   // zero pad cols 80..95
#pragma unroll
    for (int r = 0; r < 8; ++r) {
      int m = r + hi * 8;
      int oy = qy0 + (m >> 3), ox = qx0 + (m & 7);
      O16[(size_t)(oy * IW + ox) * DPAD + 80 + mcol] = (_Float16)0.0f;
    }
  }
}

// ---------------------------------------------------------------------------
// Heads: g = [sigmoid(g_raw[:4]) | softmax(g_raw[4:])], p = softmax(pi@wp+bp)
// ---------------------------------------------------------------------------
__global__ void heads_kernel(
    const float* __restrict__ gam, const float* __restrict__ pin,
    const float* __restrict__ wg, const float* __restrict__ bg,
    const float* __restrict__ wp, const float* __restrict__ bp,
    float* __restrict__ g, float* __restrict__ p)
{
  int pix = blockIdx.x * blockDim.x + threadIdx.x;
  if (pix >= NPIX) return;

  float gr[7];
#pragma unroll
  for (int i = 0; i < 7; ++i) gr[i] = bg[i];
  for (int j = 0; j < 64; ++j) {
    float gv = gam[pix * 64 + j];
#pragma unroll
    for (int i = 0; i < 7; ++i) gr[i] += gv * wg[j * 7 + i];
  }
#pragma unroll
  for (int i = 0; i < 4; ++i) g[pix * 7 + i] = 1.0f / (1.0f + expf(-gr[i]));
  float mx = fmaxf(gr[4], fmaxf(gr[5], gr[6]));
  float e0 = expf(gr[4] - mx), e1 = expf(gr[5] - mx), e2 = expf(gr[6] - mx);
  float inv = 1.0f / (e0 + e1 + e2);
  g[pix * 7 + 4] = e0 * inv; g[pix * 7 + 5] = e1 * inv; g[pix * 7 + 6] = e2 * inv;

  float pr[3];
#pragma unroll
  for (int i = 0; i < 3; ++i) pr[i] = bp[i];
  for (int j = 0; j < 8; ++j) {
    float pv = pin[pix * 8 + j];
#pragma unroll
    for (int i = 0; i < 3; ++i) pr[i] += pv * wp[j * 3 + i];
  }
  float pm = fmaxf(pr[0], fmaxf(pr[1], pr[2]));
  float q0 = expf(pr[0] - pm), q1 = expf(pr[1] - pm), q2 = expf(pr[2] - pm);
  float qi = 1.0f / (q0 + q1 + q2);
  p[pix * 3 + 0] = q0 * qi; p[pix * 3 + 1] = q1 * qi; p[pix * 3 + 2] = q2 * qi;
}

// ---------------------------------------------------------------------------
extern "C" void kernel_launch(void* const* d_in, const int* in_sizes, int n_in,
                              void* d_out, int out_size, void* d_ws,
                              size_t ws_size, hipStream_t stream)
{
  (void)in_sizes; (void)n_in; (void)out_size; (void)ws_size;

  const float* f_t   = (const float*)d_in[0];
  const float* g0    = (const float*)d_in[1];
  const float* gt    = (const float*)d_in[2];
  const float* pi_t  = (const float*)d_in[3];
  const float* wmlp1 = (const float*)d_in[4];  const float* bmlp1 = (const float*)d_in[5];
  const float* wmlp2 = (const float*)d_in[6];  const float* bmlp2 = (const float*)d_in[7];
  const float* wq1   = (const float*)d_in[8];  const float* bq1   = (const float*)d_in[9];
  const float* wk1   = (const float*)d_in[10]; const float* bk1   = (const float*)d_in[11];
  const float* wv1   = (const float*)d_in[12]; const float* bv1   = (const float*)d_in[13];
  const float* wq2   = (const float*)d_in[14]; const float* bq2   = (const float*)d_in[15];
  const float* wk2   = (const float*)d_in[16]; const float* bk2   = (const float*)d_in[17];
  const float* wv2   = (const float*)d_in[18]; const float* bv2   = (const float*)d_in[19];
  const float* w_g   = (const float*)d_in[20]; const float* b_g   = (const float*)d_in[21];
  const float* w_p   = (const float*)d_in[22]; const float* b_p   = (const float*)d_in[23];

  float* out = (float*)d_out;
  float* outG   = out;                 // 16384*7
  float* outP   = out + 114688;        // 16384*3
  float* outGam = out + 163840;        // 16384*64
  float* outPi  = out + 1212416;       // 16384*8

  char* ws = (char*)d_ws;
  _Float16* wt   = (_Float16*)ws;                  // 8 * 96*96 f16
  float*    bias = (float*)(ws + 147456);          // 8 * 96 f32
  const size_t SLOT = (size_t)NPIX * DPAD * 2;     // 3,145,728 B
  _Float16* SA = (_Float16*)(ws + 150528);
  _Float16* SB = (_Float16*)(ws + 150528 + 1 * SLOT);
  _Float16* SC = (_Float16*)(ws + 150528 + 2 * SLOT);
  _Float16* SD = (_Float16*)(ws + 150528 + 3 * SLOT);
  _Float16* SE = (_Float16*)(ws + 150528 + 4 * SLOT);

  // 1) prep
  prep_weights_kernel<<<(8 * 96 * 96 + 8 * 96 + 255) / 256, 256, 0, stream>>>(
      wmlp1, wmlp2, wq1, wk1, wv1, wq2, wk2, wv2,
      bmlp1, bmlp2, bq1, bk1, bv1, bq2, bk2, bv2, wt, bias);
  prep_inputs_kernel<<<(2 * NPIX * DPAD + 255) / 256, 256, 0, stream>>>(
      f_t, g0, gt, pi_t, SA, SB);   // SA=X1, SB=X2

  // 2) MLPs (gelu)
  gemm_kernel<<<256, 128, 0, stream>>>(SA, wt + 0 * 9216, bias + 0 * 96, SC, 1); // q
  gemm_kernel<<<256, 128, 0, stream>>>(SB, wt + 1 * 9216, bias + 1 * 96, SD, 1); // kv1

  // 3) layer-1 projections
  gemm_kernel<<<256, 128, 0, stream>>>(SC, wt + 2 * 9216, bias + 2 * 96, SA, 0); // qh1
  gemm_kernel<<<256, 128, 0, stream>>>(SD, wt + 3 * 9216, bias + 3 * 96, SB, 0); // k1
  gemm_kernel<<<256, 128, 0, stream>>>(SD, wt + 4 * 9216, bias + 4 * 96, SE, 0); // v1

  // 4) layer-1 attention -> kv2 (f16, into SD)
  natt_kernel<<<1024, 32, 0, stream>>>(SA, SB, SE, SD, nullptr, nullptr, 0);

  // 5) layer-2 projections
  gemm_kernel<<<256, 128, 0, stream>>>(SC, wt + 5 * 9216, bias + 5 * 96, SA, 0); // qh2
  gemm_kernel<<<256, 128, 0, stream>>>(SD, wt + 6 * 9216, bias + 6 * 96, SB, 0); // k2
  gemm_kernel<<<256, 128, 0, stream>>>(SD, wt + 7 * 9216, bias + 7 * 96, SE, 0); // v2

  // 6) layer-2 attention -> gamma_next / pi_next (f32, into d_out)
  natt_kernel<<<1024, 32, 0, stream>>>(SA, SB, SE, nullptr, outGam, outPi, 1);

  // 7) heads
  heads_kernel<<<(NPIX + 255) / 256, 256, 0, stream>>>(
      outGam, outPi, w_g, b_g, w_p, b_p, outG, outP);
}
